// RCAB_dense_dilated_SOCAG_15796889714785
// MI455X (gfx1250) — matrix operand order repaired and checked by reading, not compiled, not run on hardware
//
#include <hip/hip_runtime.h>
#include <hip/hip_bf16.h>
#include <math.h>

typedef __attribute__((ext_vector_type(16))) __bf16 v16bf;
typedef __attribute__((ext_vector_type(8)))  float  v8f;

#define BB 32
#define CC 256
#define HH 48
#define WW 48
#define HWSZ (HH*WW)          // 2304
#define MTOT (BB*HWSZ)        // 73728
#define NTOT ((size_t)MTOT*CC)  // 18,874,368

// ---------------------------------------------------------------------------
// Async global->LDS staging (CDNA5 GLOBAL_LOAD_ASYNC_TO_LDS, ASYNCcnt-tracked)
// ---------------------------------------------------------------------------
__device__ inline void async_copy16(void* lds_dst, const void* gsrc) {
  unsigned laddr = (unsigned)(size_t)lds_dst;   // low 32 bits of flat shared addr = LDS offset
  asm volatile("global_load_async_to_lds_b128 %0, %1, off"
               :: "v"(laddr), "v"(gsrc) : "memory");
}
__device__ inline void async_wait0() {
  asm volatile("s_wait_asynccnt 0" ::: "memory");
}

// ---------------------------------------------------------------------------
// WMMA fragment loaders (CDNA5 ISA 7.12.2 layouts, wave32)
// ---------------------------------------------------------------------------
// A: 16x32 bf16. lane(l,hi): M=l; VGPR v (pair e=2v,2v+1): K=(v&4?16:0)+hi*8+2*(v&3)+bit
__device__ inline v16bf load_a_frag(const __bf16* __restrict__ smem, int ldk) {
  int lane = threadIdx.x & 31;
  int l = lane & 15, hi = lane >> 4;
  const __bf16* row = smem + l * ldk;
  v16bf a;
#pragma unroll
  for (int v = 0; v < 8; ++v) {
    int k = ((v & 4) ? 16 : 0) + hi * 8 + 2 * (v & 3);
    a[2 * v]     = row[k];
    a[2 * v + 1] = row[k + 1];
  }
  return a;
}
// B: 32x16 bf16, stored transposed in LDS as [N][K]. lane(l,hi): N=l; K=(e&8?16:0)+hi*8+(e&7)
__device__ inline v16bf load_b_frag(const __bf16* __restrict__ smem, int ldk) {
  int lane = threadIdx.x & 31;
  int l = lane & 15, hi = lane >> 4;
  const __bf16* col = smem + l * ldk + hi * 8;
  v16bf b;
#pragma unroll
  for (int e = 0; e < 16; ++e) b[e] = col[(e & 8) * 2 + (e & 7)];
  return b;
}

// ---------------------------------------------------------------------------
// Layout / conversion kernels
// ---------------------------------------------------------------------------
__global__ void k_nchw_to_nhwc_bf16(const float* __restrict__ src, __bf16* __restrict__ dst) {
  size_t idx = (size_t)blockIdx.x * blockDim.x + threadIdx.x;
  if (idx >= NTOT) return;
  int c = (int)(idx % CC);
  int s = (int)(idx / CC);
  int b = s / HWSZ, p = s % HWSZ;
  dst[idx] = (__bf16)src[((size_t)(b * CC + c)) * HWSZ + p];
}

// w (O,I,3,3) fp32 -> wt[tap][I][O] bf16
__global__ void k_wreorg(const float* __restrict__ w, __bf16* __restrict__ wt, int O, int I) {
  int idx = blockIdx.x * blockDim.x + threadIdx.x;
  int total = O * I * 9;
  if (idx >= total) return;
  int tap = idx % 9;
  int i = (idx / 9) % I;
  int o = idx / (9 * I);
  wt[((size_t)tap * I + i) * O + o] = (__bf16)w[((size_t)o * I + i) * 9 + tap];
}

__global__ void k_f2bf(const float* __restrict__ src, __bf16* __restrict__ dst, size_t n) {
  size_t idx = (size_t)blockIdx.x * blockDim.x + threadIdx.x;
  if (idx < n) dst[idx] = (__bf16)src[idx];
}

// ---------------------------------------------------------------------------
// Implicit-GEMM 3x3 SAME conv, NHWC bf16 in, weights [tap][Cin][Cout] bf16
// Block: 128 threads (4 waves, 2x2), tile M=64 (spatial) x N=64 (Cout), K step 32.
// A tile staged via GLOBAL_LOAD_ASYNC_TO_LDS_B128 (zero-filled at halo).
// ---------------------------------------------------------------------------
template <int BIAS, int RELU, int WF32, int WBF>
__global__ void k_conv3x3(const __bf16* __restrict__ A, const __bf16* __restrict__ Wt,
                          const float* __restrict__ bias,
                          float* __restrict__ outF, __bf16* __restrict__ outB,
                          int Cin, int Cout) {
  const int LDA = 40, LDB = 40;
  __shared__ __bf16 As[64 * 40];
  __shared__ __bf16 Bs[64 * 40];
  int tid  = threadIdx.x;
  int wave = tid >> 5, lane = tid & 31;
  int wm = wave >> 1, wn = wave & 1;
  int m0 = blockIdx.y * 64;
  int n0 = blockIdx.x * 64;
  v8f acc[2][2] = {};

  for (int tap = 0; tap < 9; ++tap) {
    int dy = tap / 3 - 1, dx = tap % 3 - 1;
    for (int kc = 0; kc < Cin; kc += 32) {
      // ---- stage A tile (64 rows x 32 ch): async copy, zero-fill OOB taps
#pragma unroll
      for (int rep = 0; rep < 2; ++rep) {
        int slot = tid + rep * 128;       // 0..255
        int r = slot >> 2, g = slot & 3;  // row, 8-elem segment
        int s = m0 + r;
        int b = s / HWSZ, p = s - b * HWSZ;
        int y = p / WW, x = p - y * WW;
        int ys = y + dy, xs = x + dx;
        __bf16* lp = As + r * LDA + g * 8;
        if (ys >= 0 && ys < HH && xs >= 0 && xs < WW) {
          async_copy16(lp, A + ((size_t)(b * HWSZ + ys * WW + xs)) * Cin + kc + g * 8);
        } else {
          *(uint4*)lp = make_uint4(0u, 0u, 0u, 0u);
        }
      }
      // ---- stage B tile (32 k x 64 n), coalesced read, transpose-scatter to [n][k]
#pragma unroll
      for (int rep = 0; rep < 2; ++rep) {
        int slot = tid + rep * 128;
        int k = slot >> 3, nseg = slot & 7;
        const __bf16* gp = Wt + ((size_t)tap * Cin + kc + k) * Cout + n0 + nseg * 8;
        __bf16 tmp[8];
        *(uint4*)tmp = *(const uint4*)gp;
#pragma unroll
        for (int e = 0; e < 8; ++e) Bs[(nseg * 8 + e) * LDB + k] = tmp[e];
      }
      async_wait0();
      __syncthreads();
      v16bf a0 = load_a_frag(As + (wm * 32) * LDA, LDA);
      v16bf a1 = load_a_frag(As + (wm * 32 + 16) * LDA, LDA);
      v16bf b0 = load_b_frag(Bs + (wn * 32) * LDB, LDB);
      v16bf b1 = load_b_frag(Bs + (wn * 32 + 16) * LDB, LDB);
      acc[0][0] = __builtin_amdgcn_wmma_f32_16x16x32_bf16(false, a0, false, b0, (short)0, acc[0][0], false, false);
      acc[0][1] = __builtin_amdgcn_wmma_f32_16x16x32_bf16(false, a0, false, b1, (short)0, acc[0][1], false, false);
      acc[1][0] = __builtin_amdgcn_wmma_f32_16x16x32_bf16(false, a1, false, b0, (short)0, acc[1][0], false, false);
      acc[1][1] = __builtin_amdgcn_wmma_f32_16x16x32_bf16(false, a1, false, b1, (short)0, acc[1][1], false, false);
      __syncthreads();
    }
  }
  // ---- epilogue
  int l = lane & 15, hi = lane >> 4;
#pragma unroll
  for (int tm = 0; tm < 2; ++tm)
#pragma unroll
    for (int tn = 0; tn < 2; ++tn) {
      int col = n0 + wn * 32 + tn * 16 + l;
      float bv = BIAS ? bias[col] : 0.f;
#pragma unroll
      for (int r = 0; r < 8; ++r) {
        int row = m0 + wm * 32 + tm * 16 + hi * 8 + r;
        float vv = acc[tm][tn][r] + bv;
        if (RELU) vv = fmaxf(vv, 0.f);
        size_t o = (size_t)row * Cout + col;
        if (WF32) outF[o] = vv;
        if (WBF)  outB[o] = (__bf16)vv;
      }
    }
}

// ---------------------------------------------------------------------------
// Batched GEMM: C[bz](MxN,f32) = alpha * A[bz](MxK bf16) * B[bz](KxN bf16)
// A tile staged via async-to-LDS. M,N multiples of 64; K multiple of 32.
// ---------------------------------------------------------------------------
__global__ void k_gemm_bf16(const __bf16* __restrict__ Ag, const __bf16* __restrict__ Bg,
                            float* __restrict__ Cf, int Mm, int Nn, int Kk, float alpha) {
  const int LDA = 40, LDB = 40;
  __shared__ __bf16 As[64 * 40];
  __shared__ __bf16 Bs[64 * 40];
  int tid  = threadIdx.x;
  int wave = tid >> 5, lane = tid & 31;
  int wm = wave >> 1, wn = wave & 1;
  int m0 = blockIdx.y * 64;
  int n0 = blockIdx.x * 64;
  int bz = blockIdx.z;
  const __bf16* A = Ag + (size_t)bz * Mm * Kk;
  const __bf16* B = Bg + (size_t)bz * Kk * Nn;
  v8f acc[2][2] = {};

  for (int kc = 0; kc < Kk; kc += 32) {
#pragma unroll
    for (int rep = 0; rep < 2; ++rep) {
      int slot = tid + rep * 128;
      int r = slot >> 2, g = slot & 3;
      async_copy16(As + r * LDA + g * 8, A + (size_t)(m0 + r) * Kk + kc + g * 8);
    }
#pragma unroll
    for (int rep = 0; rep < 2; ++rep) {
      int slot = tid + rep * 128;
      int k = slot >> 3, nseg = slot & 7;
      __bf16 tmp[8];
      *(uint4*)tmp = *(const uint4*)(B + (size_t)(kc + k) * Nn + n0 + nseg * 8);
#pragma unroll
      for (int e = 0; e < 8; ++e) Bs[(nseg * 8 + e) * LDB + k] = tmp[e];
    }
    async_wait0();
    __syncthreads();
    v16bf a0 = load_a_frag(As + (wm * 32) * LDA, LDA);
    v16bf a1 = load_a_frag(As + (wm * 32 + 16) * LDA, LDA);
    v16bf b0 = load_b_frag(Bs + (wn * 32) * LDB, LDB);
    v16bf b1 = load_b_frag(Bs + (wn * 32 + 16) * LDB, LDB);
    acc[0][0] = __builtin_amdgcn_wmma_f32_16x16x32_bf16(false, a0, false, b0, (short)0, acc[0][0], false, false);
    acc[0][1] = __builtin_amdgcn_wmma_f32_16x16x32_bf16(false, a0, false, b1, (short)0, acc[0][1], false, false);
    acc[1][0] = __builtin_amdgcn_wmma_f32_16x16x32_bf16(false, a1, false, b0, (short)0, acc[1][0], false, false);
    acc[1][1] = __builtin_amdgcn_wmma_f32_16x16x32_bf16(false, a1, false, b1, (short)0, acc[1][1], false, false);
    __syncthreads();
  }
  float* C = Cf + (size_t)bz * Mm * Nn;
  int l = lane & 15, hi = lane >> 4;
#pragma unroll
  for (int tm = 0; tm < 2; ++tm)
#pragma unroll
    for (int tn = 0; tn < 2; ++tn) {
      int col = n0 + wn * 32 + tn * 16 + l;
#pragma unroll
      for (int r = 0; r < 8; ++r) {
        int row = m0 + wm * 32 + tm * 16 + hi * 8 + r;
        C[(size_t)row * Nn + col] = alpha * acc[tm][tn][r];
      }
    }
}

// ---------------------------------------------------------------------------
// Deformable sampling with torch .view(b*c,h,w,2) offset scrambling.
// img: fp32, NCHW (template=1) or NHWC (template=0). off: bf16 NHWC (b,s,2C).
// out: bf16 NHWC.
// ---------------------------------------------------------------------------
template <int IMG_NCHW>
__global__ void k_deform(const float* __restrict__ img, const __bf16* __restrict__ offb,
                         __bf16* __restrict__ outb) {
  size_t idx = (size_t)blockIdx.x * blockDim.x + threadIdx.x;
  if (idx >= NTOT) return;
  int c = (int)(idx % CC);
  int s = (int)(idx / CC);
  int b = s / HWSZ, p = s % HWSZ;
  int y = p / WW, x = p % WW;
  float oy, ox;
  {
    int q = p * 2;
    int ch2 = 2 * c + (q >= HWSZ ? 1 : 0);
    int ss  = q >= HWSZ ? q - HWSZ : q;
    oy = (float)offb[((size_t)b * HWSZ + ss) * (2 * CC) + ch2];
  }
  {
    int q = p * 2 + 1;
    int ch2 = 2 * c + (q >= HWSZ ? 1 : 0);
    int ss  = q >= HWSZ ? q - HWSZ : q;
    ox = (float)offb[((size_t)b * HWSZ + ss) * (2 * CC) + ch2];
  }
  float fy = fminf(fmaxf((float)y + oy, 0.f), (float)(HH - 1));
  float fx = fminf(fmaxf((float)x + ox, 0.f), (float)(WW - 1));
  int y0 = (int)floorf(fy), x0 = (int)floorf(fx);
  int y1 = min(y0 + 1, HH - 1), x1 = min(x0 + 1, WW - 1);
  float wy = fy - (float)y0, wx = fx - (float)x0;
  auto ld = [&](int yy, int xx) -> float {
    if (IMG_NCHW) return img[((size_t)(b * CC + c)) * HWSZ + yy * WW + xx];
    else          return img[((size_t)(b * HWSZ) + yy * WW + xx) * CC + c];
  };
  float v00 = ld(y0, x0), v01 = ld(y0, x1), v10 = ld(y1, x0), v11 = ld(y1, x1);
  float r = v00 * (1 - wy) * (1 - wx) + v01 * (1 - wy) * wx +
            v10 * wy * (1 - wx) + v11 * wy * wx;
  outb[idx] = (__bf16)r;
}

// ---------------------------------------------------------------------------
// SOCA helper kernels
// ---------------------------------------------------------------------------
__global__ void k_chan_mean(const float* __restrict__ a, float* __restrict__ meanc) {
  int b = blockIdx.x, c = threadIdx.x;
  float acc = 0.f;
  for (int s = 0; s < HWSZ; ++s) acc += a[((size_t)b * HWSZ + s) * CC + c];
  meanc[b * CC + c] = acc / (float)HWSZ;
}

__global__ void k_center(const float* __restrict__ a, const float* __restrict__ meanc,
                         __bf16* __restrict__ XcT, __bf16* __restrict__ XcC) {
  size_t idx = (size_t)blockIdx.x * blockDim.x + threadIdx.x;
  if (idx >= NTOT) return;
  int c = (int)(idx % CC);
  int s = (int)(idx / CC);
  int b = s / HWSZ, p = s % HWSZ;
  float v = a[idx] - meanc[b * CC + c];
  XcT[idx] = (__bf16)v;                                    // (b, m, c)
  XcC[((size_t)b * CC + c) * HWSZ + p] = (__bf16)v;        // (b, c, m)
}

__global__ void k_trace(const float* __restrict__ cov, float* __restrict__ normA) {
  __shared__ float red[CC];
  int b = blockIdx.x, t = threadIdx.x;
  red[t] = cov[((size_t)b * CC + t) * CC + t];
  __syncthreads();
  for (int st = CC / 2; st > 0; st >>= 1) {
    if (t < st) red[t] += red[t + st];
    __syncthreads();
  }
  if (t == 0) normA[b] = red[0];
}

__global__ void k_ns_init(const float* __restrict__ cov, const float* __restrict__ normA,
                          float* __restrict__ Yf, __bf16* __restrict__ Ybf,
                          __bf16* __restrict__ Zbf) {
  size_t idx = (size_t)blockIdx.x * blockDim.x + threadIdx.x;
  size_t total = (size_t)BB * CC * CC;
  if (idx >= total) return;
  int b = (int)(idx / (CC * CC));
  int r = (int)((idx / CC) % CC);
  int c = (int)(idx % CC);
  float y = cov[idx] / normA[b];
  Yf[idx] = y;
  Ybf[idx] = (__bf16)y;
  Zbf[idx] = (__bf16)((r == c) ? 1.f : 0.f);
}

__global__ void k_ns_t(float* __restrict__ Tf, __bf16* __restrict__ Tbf) {
  size_t idx = (size_t)blockIdx.x * blockDim.x + threadIdx.x;
  size_t total = (size_t)BB * CC * CC;
  if (idx >= total) return;
  int r = (int)((idx / CC) % CC);
  int c = (int)(idx % CC);
  float t = 1.5f * ((r == c) ? 1.f : 0.f) - 0.5f * Tf[idx];
  Tf[idx] = t;
  Tbf[idx] = (__bf16)t;
}

__global__ void k_colmean(const float* __restrict__ Yf, const float* __restrict__ normA,
                          float* __restrict__ v) {
  int b = blockIdx.x, j = threadIdx.x;
  float acc = 0.f;
  for (int i = 0; i < CC; ++i) acc += Yf[((size_t)b * CC + i) * CC + j];
  v[b * CC + j] = acc * sqrtf(normA[b]) / (float)CC;
}

__global__ void k_mlp(const float* __restrict__ v, const float* __restrict__ w1,
                      const float* __restrict__ b1, const float* __restrict__ w2,
                      const float* __restrict__ b2, float* __restrict__ attn) {
  __shared__ float sv[CC];
  __shared__ float sh[CC / 8];
  int b = blockIdx.x, t = threadIdx.x;
  sv[t] = v[b * CC + t];
  __syncthreads();
  if (t < CC / 8) {
    float acc = b1[t];
    for (int c = 0; c < CC; ++c) acc += sv[c] * w1[t * CC + c];
    sh[t] = fmaxf(acc, 0.f);
  }
  __syncthreads();
  float acc = b2[t];
  for (int j = 0; j < CC / 8; ++j) acc += sh[j] * w2[t * (CC / 8) + j];
  attn[b * CC + t] = 1.f / (1.f + expf(-acc));
}

__global__ void k_final(const float* __restrict__ a4, const float* __restrict__ attn,
                        const float* __restrict__ x, float* __restrict__ out) {
  size_t idx = (size_t)blockIdx.x * blockDim.x + threadIdx.x;
  if (idx >= NTOT) return;
  int p = (int)(idx % HWSZ);
  int bc = (int)(idx / HWSZ);
  int c = bc % CC, b = bc / CC;
  float av = a4[((size_t)b * HWSZ + p) * CC + c];
  out[idx] = attn[b * CC + c] * av + x[idx];
}

// ---------------------------------------------------------------------------
// Host glue
// ---------------------------------------------------------------------------
extern "C" void kernel_launch(void* const* d_in, const int* in_sizes, int n_in,
                              void* d_out, int out_size, void* d_ws, size_t ws_size,
                              hipStream_t stream) {
  const float* x      = (const float*)d_in[0];
  const float* off_w1 = (const float*)d_in[1];
  const float* off_w2 = (const float*)d_in[2];
  const float* c1w    = (const float*)d_in[3];
  const float* c1b    = (const float*)d_in[4];
  const float* c2w    = (const float*)d_in[5];
  const float* c2b    = (const float*)d_in[6];
  const float* caw1   = (const float*)d_in[7];
  const float* cab1   = (const float*)d_in[8];
  const float* caw2   = (const float*)d_in[9];
  const float* cab2   = (const float*)d_in[10];
  float* out = (float*)d_out;

  char* ws = (char*)d_ws;
  size_t cur = 0;
  auto alloc = [&](size_t bytes) -> void* {
    void* p = (void*)(ws + cur);
    cur += (bytes + 255) & ~(size_t)255;
    return p;
  };
  __bf16* xbf  = (__bf16*)alloc(NTOT * 2);                       // later XcT
  __bf16* wto1 = (__bf16*)alloc((size_t)9 * CC * 2 * CC * 2);
  __bf16* wto2 = (__bf16*)alloc((size_t)9 * CC * 2 * CC * 2);
  __bf16* wtc1 = (__bf16*)alloc((size_t)9 * CC * CC * 2);
  __bf16* wtc2 = (__bf16*)alloc((size_t)9 * CC * CC * 2);
  __bf16* offb = (__bf16*)alloc((size_t)MTOT * 2 * CC * 2);      // off1/off2
  __bf16* abf  = (__bf16*)alloc(NTOT * 2);                       // a1bf/a3bf, later XcC
  __bf16* abf2 = (__bf16*)alloc(NTOT * 2);                       // a2bf
  float*  act2 = (float*)alloc(NTOT * 4);                        // a2 then a4 (NHWC f32)
  float*  covb = (float*)alloc((size_t)BB * CC * CC * 4);
  float*  Yf   = (float*)alloc((size_t)BB * CC * CC * 4);
  float*  Zf   = (float*)alloc((size_t)BB * CC * CC * 4);
  float*  Tf   = (float*)alloc((size_t)BB * CC * CC * 4);
  __bf16* Ybf  = (__bf16*)alloc((size_t)BB * CC * CC * 2);
  __bf16* Zbf  = (__bf16*)alloc((size_t)BB * CC * CC * 2);
  __bf16* Tbf  = (__bf16*)alloc((size_t)BB * CC * CC * 2);
  float*  meanc = (float*)alloc((size_t)BB * CC * 4);
  float*  normA = (float*)alloc((size_t)BB * 4);
  float*  vvec  = (float*)alloc((size_t)BB * CC * 4);
  float*  attn  = (float*)alloc((size_t)BB * CC * 4);
  __bf16* XcT = xbf;
  __bf16* XcC = abf;

  const int TB = 256;
  dim3 gElem((unsigned)((NTOT + TB - 1) / TB));
  dim3 gNS((unsigned)(((size_t)BB * CC * CC + TB - 1) / TB));

  // 1. x -> NHWC bf16; weight repacks
  k_nchw_to_nhwc_bf16<<<gElem, TB, 0, stream>>>(x, xbf);
  k_wreorg<<<(2 * CC * CC * 9 + TB - 1) / TB, TB, 0, stream>>>(off_w1, wto1, 2 * CC, CC);
  k_wreorg<<<(2 * CC * CC * 9 + TB - 1) / TB, TB, 0, stream>>>(off_w2, wto2, 2 * CC, CC);
  k_wreorg<<<(CC * CC * 9 + TB - 1) / TB, TB, 0, stream>>>(c1w, wtc1, CC, CC);
  k_wreorg<<<(CC * CC * 9 + TB - 1) / TB, TB, 0, stream>>>(c2w, wtc2, CC, CC);

  dim3 gConvOff(2 * CC / 64, MTOT / 64), gConv(CC / 64, MTOT / 64);
  // 2. off1 = conv(x); sample
  k_conv3x3<0, 0, 0, 1><<<gConvOff, 128, 0, stream>>>(xbf, wto1, nullptr, nullptr, offb, CC, 2 * CC);
  k_deform<1><<<gElem, TB, 0, stream>>>(x, offb, abf);
  // 3. a2 = relu(conv1(a1)+b1) -> f32 + bf16
  k_conv3x3<1, 1, 1, 1><<<gConv, 128, 0, stream>>>(abf, wtc1, c1b, act2, abf2, CC, CC);
  // 4. off2 = conv(a2); sample from a2 (NHWC f32)
  k_conv3x3<0, 0, 0, 1><<<gConvOff, 128, 0, stream>>>(abf2, wto2, nullptr, nullptr, offb, CC, 2 * CC);
  k_deform<0><<<gElem, TB, 0, stream>>>(act2, offb, abf);
  // 5. a4 = conv2(a3)+b2 -> f32 NHWC (overwrites act2)
  k_conv3x3<1, 0, 1, 0><<<gConv, 128, 0, stream>>>(abf, wtc2, c2b, act2, nullptr, CC, CC);

  // 6. SOCA: center, covariance, Newton-Schulz
  k_chan_mean<<<BB, CC, 0, stream>>>(act2, meanc);
  k_center<<<gElem, TB, 0, stream>>>(act2, meanc, XcT, XcC);
  dim3 gCov(CC / 64, CC / 64, BB);
  k_gemm_bf16<<<gCov, 128, 0, stream>>>(XcC, XcT, covb, CC, CC, HWSZ, 1.f / (float)HWSZ);
  k_trace<<<BB, CC, 0, stream>>>(covb, normA);
  k_ns_init<<<gNS, TB, 0, stream>>>(covb, normA, Yf, Ybf, Zbf);
  size_t nsN = (size_t)BB * CC * CC;
  for (int it = 0; it < 5; ++it) {
    k_gemm_bf16<<<gCov, 128, 0, stream>>>(Zbf, Ybf, Tf, CC, CC, CC, 1.f);  // P = Z@Y
    k_ns_t<<<gNS, TB, 0, stream>>>(Tf, Tbf);                                // T = 1.5I-0.5P
    k_gemm_bf16<<<gCov, 128, 0, stream>>>(Ybf, Tbf, Yf, CC, CC, CC, 1.f);  // Y = Y@T
    k_gemm_bf16<<<gCov, 128, 0, stream>>>(Tbf, Zbf, Zf, CC, CC, CC, 1.f);  // Z = T@Z
    k_f2bf<<<gNS, TB, 0, stream>>>(Yf, Ybf, nsN);
    k_f2bf<<<gNS, TB, 0, stream>>>(Zf, Zbf, nsN);
  }
  // 7. v = (sqrt(normA)/C) * colsum(Y); MLP; scale + residual
  k_colmean<<<BB, CC, 0, stream>>>(Yf, normA, vvec);
  k_mlp<<<BB, CC, 0, stream>>>(vvec, caw1, cab1, caw2, cab2, attn);
  k_final<<<gElem, TB, 0, stream>>>(act2, attn, x, out);
}
